// GNN_embedder_90374701843011
// MI455X (gfx1250) — matrix operand order repaired and checked by reading, not compiled, not run on hardware
//
#include <hip/hip_runtime.h>
#include <hip/hip_bf16.h>

typedef __attribute__((ext_vector_type(16))) _Float16 v16h;
typedef __attribute__((ext_vector_type(8)))  float    v8f;
typedef int gvec4i __attribute__((vector_size(16)));   // async builtin pointee type

#define IN_C   8
#define OUT_C  16
#define EDGE_F 8
#define HEADS  4
#define HID    64
#define GDIM   256   // HEADS*HID
#define FDIM   128   // FINAL
#define NEG_SLOPE 0.2f

union HalfPack { _Float16 h[2]; unsigned int u; };

// ---------------------------------------------------------------- utilities
__device__ __forceinline__ void atomicMaxF(float* addr, float val) {
  unsigned int* ua = (unsigned int*)addr;
  unsigned int cur = __float_as_uint(*addr);
  while (__uint_as_float(cur) < val) {
    unsigned int prev = atomicCAS(ua, cur, __float_as_uint(val));
    if (prev == cur) break;
    cur = prev;
  }
}

__device__ __forceinline__ float leaky(float v) {
  return v > 0.f ? v : NEG_SLOPE * v;
}

template <int O>
__device__ __forceinline__ void pack4_relu(v16h& a, float4 v, float4 b) {
  float t;
  t = v.x + b.x; a[O + 0] = (_Float16)(t > 0.f ? t : 0.f);
  t = v.y + b.y; a[O + 1] = (_Float16)(t > 0.f ? t : 0.f);
  t = v.z + b.z; a[O + 2] = (_Float16)(t > 0.f ? t : 0.f);
  t = v.w + b.w; a[O + 3] = (_Float16)(t > 0.f ? t : 0.f);
}

__global__ void init_emax_kernel(float* __restrict__ emax, int n) {
  int i = blockIdx.x * blockDim.x + threadIdx.x;
  if (i < n) emax[i] = -1e30f;
}

// ------------------------------------------------- NNConv edge messages (add)
__global__ void ecc_edge_kernel(const float* __restrict__ x,
                                const int*   __restrict__ ei,
                                const float* __restrict__ ea,
                                const float* __restrict__ mlp_w,
                                const float* __restrict__ mlp_b,
                                float* __restrict__ agg, int E) {
  __shared__ __align__(16) float sw[EDGE_F * IN_C * OUT_C];   // 4 KB
  __shared__ __align__(16) float sb[IN_C * OUT_C];            // 512 B

#if __has_builtin(__builtin_amdgcn_global_load_async_to_lds_b128)
  {
    int t = threadIdx.x;                       // blockDim.x == 256
    // 256 lanes x 16B = 4KB of mlp_w, async DMA into LDS
    __builtin_amdgcn_global_load_async_to_lds_b128(
        (gvec4i*)((const char*)mlp_w + t * 16),
        (gvec4i*)((char*)sw + t * 16), 0, 0);
    if (t < 32)
      __builtin_amdgcn_global_load_async_to_lds_b128(
          (gvec4i*)((const char*)mlp_b + t * 16),
          (gvec4i*)((char*)sb + t * 16), 0, 0);
#if __has_builtin(__builtin_amdgcn_s_wait_asynccnt)
    __builtin_amdgcn_s_wait_asynccnt(0);
#else
    asm volatile("s_wait_asynccnt 0" ::: "memory");
#endif
  }
#else
  for (int i = threadIdx.x; i < EDGE_F * IN_C * OUT_C; i += blockDim.x) sw[i] = mlp_w[i];
  for (int i = threadIdx.x; i < IN_C * OUT_C; i += blockDim.x) sb[i] = mlp_b[i];
#endif
  __syncthreads();

  int e = blockIdx.x * blockDim.x + threadIdx.x;
  if (e >= E) return;
  int src = ei[e];
  int dst = ei[E + e];

  float xv[IN_C], eav[EDGE_F];
#pragma unroll
  for (int i = 0; i < IN_C; ++i) xv[i] = x[src * IN_C + i];
#pragma unroll
  for (int f = 0; f < EDGE_F; ++f) eav[f] = ea[e * EDGE_F + f];

  for (int o = 0; o < OUT_C; ++o) {
    float acc = 0.f;
#pragma unroll
    for (int i = 0; i < IN_C; ++i) {
      float w = sb[i * OUT_C + o];
#pragma unroll
      for (int f = 0; f < EDGE_F; ++f)
        w += eav[f] * sw[f * (IN_C * OUT_C) + i * OUT_C + o];
      acc += xv[i] * w;
    }
    atomicAdd(&agg[dst * OUT_C + o], acc);
  }
}

// ------------------------------------------- h = relu(x@root + agg + bias)
__global__ void ecc_node_kernel(const float* __restrict__ x,
                                const float* __restrict__ ecc_root,
                                const float* __restrict__ ecc_bias,
                                const float* __restrict__ agg,
                                float* __restrict__ hbuf, int N) {
  int nid = blockIdx.x * blockDim.x + threadIdx.x;
  if (nid >= N) return;
  float xv[IN_C];
#pragma unroll
  for (int i = 0; i < IN_C; ++i) xv[i] = x[nid * IN_C + i];
  for (int o = 0; o < OUT_C; ++o) {
    float acc = ecc_bias[o] + agg[nid * OUT_C + o];
#pragma unroll
    for (int i = 0; i < IN_C; ++i) acc += xv[i] * ecc_root[i * OUT_C + o];
    hbuf[nid * OUT_C + o] = acc > 0.f ? acc : 0.f;
  }
}

// ---------------------------------------- g = h @ gat_lin  (WMMA f16, K=16→32)
// Whole-tile kernel: processes nTiles full 16-node tiles, no store guards.
// B staged in LDS pre-swizzled to the per-lane WMMA fragment layout:
// dword index = (ct*32 + lane)*8 + j holds {B[k,n], B[k+1,n]},
// k = khalf*8 + 2j for j<4 (zeros for j>=4 == K padding), n = lane&15.
__global__ void gat_gemm_kernel(const float* __restrict__ hbuf,     // [N,16]
                                const float* __restrict__ gat_lin,  // [16,256]
                                float* __restrict__ g, int nTiles) {
  __shared__ __align__(32) unsigned int ldsb[(GDIM / 16) * 32 * 8];  // 16 KB
  for (int idx = threadIdx.x; idx < (GDIM / 16) * 32 * 8; idx += blockDim.x) {
    int j = idx & 7, lane = (idx >> 3) & 31, ct = idx >> 8;
    int khalf = lane >> 4, n = lane & 15;
    HalfPack p;
    p.u = 0u;
    if (j < 4) {
      p.h[0] = (_Float16)gat_lin[(khalf * 8 + 2 * j) * GDIM + ct * 16 + n];
      p.h[1] = (_Float16)gat_lin[(khalf * 8 + 2 * j + 1) * GDIM + ct * 16 + n];
    }
    ldsb[idx] = p.u;
  }
  __syncthreads();

  int wave = threadIdx.x >> 5;
  int lane = threadIdx.x & 31;
  int tile = blockIdx.x * 8 + wave;
  if (tile >= nTiles) return;          // wave-uniform
  int nodeBase = tile * 16;

  int m = lane & 15;
  int khalf = lane >> 4;
  int n = m;
  int node = nodeBase + m;

  // A: 16x32 f16 (K=16 real, upper 16 zero-padded); 2x b128 loads per lane
  const float4* hr = (const float4*)(hbuf + (size_t)node * OUT_C);
  float4 h0 = hr[khalf * 2];
  float4 h1 = hr[khalf * 2 + 1];
  v16h a;
#pragma unroll
  for (int e = 8; e < 16; ++e) a[e] = (_Float16)0.0f;
  a[0] = (_Float16)h0.x; a[1] = (_Float16)h0.y;
  a[2] = (_Float16)h0.z; a[3] = (_Float16)h0.w;
  a[4] = (_Float16)h1.x; a[5] = (_Float16)h1.y;
  a[6] = (_Float16)h1.z; a[7] = (_Float16)h1.w;

  for (int ct = 0; ct < GDIM / 16; ++ct) {
    int col0 = ct * 16;
    v16h b = *(const v16h*)&ldsb[(ct * 32 + lane) * 8];
    v8f c;
#pragma unroll
    for (int r = 0; r < 8; ++r) c[r] = 0.f;
    c = __builtin_amdgcn_wmma_f32_16x16x32_f16(false, a, false, b,
                                               (short)0, c, false, false);
#pragma unroll
    for (int r = 0; r < 8; ++r)
      g[(size_t)(nodeBase + r + khalf * 8) * GDIM + col0 + n] = c[r];
  }
}

// scalar tail for the last N%16 nodes (not launched when N%16==0)
__global__ void gat_tail_kernel(const float* __restrict__ hbuf,
                                const float* __restrict__ gat_lin,
                                float* __restrict__ g, int startNode, int N) {
  int idx = blockIdx.x * blockDim.x + threadIdx.x;
  int total = (N - startNode) * GDIM;
  if (idx >= total) return;
  int node = startNode + idx / GDIM;
  int col = idx % GDIM;
  float acc = 0.f;
#pragma unroll
  for (int k = 0; k < OUT_C; ++k)
    acc += hbuf[node * OUT_C + k] * gat_lin[k * GDIM + col];
  g[(size_t)node * GDIM + col] = acc;
}

// ----------------------------------------------- per (node,head) attn scores
__global__ void att_score_kernel(const float* __restrict__ g,
                                 const float* __restrict__ att_src,
                                 const float* __restrict__ att_dst,
                                 float* __restrict__ asrc,
                                 float* __restrict__ adst, int N) {
  int t = blockIdx.x * blockDim.x + threadIdx.x;
  if (t >= N * HEADS) return;
  int nid = t / HEADS, hh = t % HEADS;
  const float* gr = g + (size_t)nid * GDIM + hh * HID;
  float s1 = 0.f, s2 = 0.f;
  for (int d = 0; d < HID; ++d) {
    float gv = gr[d];
    s1 += gv * att_src[hh * HID + d];
    s2 += gv * att_dst[hh * HID + d];
  }
  asrc[t] = s1;
  adst[t] = s2;
}

// ----------------------------------------------------------- segment max
__global__ void edge_max_kernel(const int* __restrict__ ei,
                                const float* __restrict__ asrc,
                                const float* __restrict__ adst,
                                float* __restrict__ emax, int E) {
  int e = blockIdx.x * blockDim.x + threadIdx.x;
  if (e >= E) return;
  int src = ei[e], dst = ei[E + e];
#pragma unroll
  for (int hh = 0; hh < HEADS; ++hh) {
    float v = leaky(asrc[src * HEADS + hh] + adst[dst * HEADS + hh]);
    atomicMaxF(&emax[dst * HEADS + hh], v);
  }
}

// ----------------------------------------------------------- softmax denom
__global__ void edge_denom_kernel(const int* __restrict__ ei,
                                  const float* __restrict__ asrc,
                                  const float* __restrict__ adst,
                                  const float* __restrict__ emax,
                                  float* __restrict__ denom, int E) {
  int e = blockIdx.x * blockDim.x + threadIdx.x;
  if (e >= E) return;
  int src = ei[e], dst = ei[E + e];
#pragma unroll
  for (int hh = 0; hh < HEADS; ++hh) {
    float v = leaky(asrc[src * HEADS + hh] + adst[dst * HEADS + hh]);
    atomicAdd(&denom[dst * HEADS + hh], __expf(v - emax[dst * HEADS + hh]));
  }
}

// --------------------------------------- out_acc[dst] += alpha * g[src]
// 64 threads per edge; thread t handles channel t of each of the 4 heads.
__global__ void edge_accum_kernel(const int* __restrict__ ei,
                                  const float* __restrict__ asrc,
                                  const float* __restrict__ adst,
                                  const float* __restrict__ emax,
                                  const float* __restrict__ denom,
                                  const float* __restrict__ g,
                                  float* __restrict__ outac, int E) {
  long long tid = (long long)blockIdx.x * blockDim.x + threadIdx.x;
  int e = (int)(tid >> 6);
  int t = (int)(tid & 63);
  if (e >= E) return;
  int src = ei[e], dst = ei[E + e];
#pragma unroll
  for (int hh = 0; hh < HEADS; ++hh) {
    float v = leaky(asrc[src * HEADS + hh] + adst[dst * HEADS + hh]);
    float ex = __expf(v - emax[dst * HEADS + hh]);
    float al = ex / (denom[dst * HEADS + hh] + 1e-16f);
    int c = hh * HID + t;
    atomicAdd(&outac[(size_t)dst * GDIM + c], al * g[(size_t)src * GDIM + c]);
  }
}

// ------------------ y = relu(out_acc + gat_bias) @ fc_w + fc_b   (WMMA f16)
// Whole-tile kernel: nTiles full 16-node tiles, no store guards.
// fc_w staged in LDS pre-swizzled per-lane fragments:
// dword index = ((ct*8 + ks)*32 + lane)*8 + j holds {W[k,col], W[k+1,col]},
// k = ks*32 + (j>>2)*16 + khalf*8 + 2(j&3), col = ct*16 + (lane&15).
__global__ void final_gemm_kernel(const float* __restrict__ outac,   // [N,256]
                                  const float* __restrict__ gat_bias,// [256]
                                  const float* __restrict__ fc_w,    // [256,128]
                                  const float* __restrict__ fc_b,    // [128]
                                  float* __restrict__ y, int nTiles) {
  __shared__ __align__(32) unsigned int ldsb[(FDIM / 16) * 8 * 32 * 8]; // 64 KB
  for (int idx = threadIdx.x; idx < (FDIM / 16) * 8 * 32 * 8; idx += blockDim.x) {
    int j = idx & 7, lane = (idx >> 3) & 31, ks = (idx >> 8) & 7, ct = idx >> 11;
    int khalf = lane >> 4, n = lane & 15;
    int k = ks * 32 + (j >> 2) * 16 + khalf * 8 + 2 * (j & 3);
    HalfPack p;
    p.h[0] = (_Float16)fc_w[k * FDIM + ct * 16 + n];
    p.h[1] = (_Float16)fc_w[(k + 1) * FDIM + ct * 16 + n];
    ldsb[idx] = p.u;
  }
  __syncthreads();

  int wave = threadIdx.x >> 5;
  int lane = threadIdx.x & 31;
  int tile = blockIdx.x * 8 + wave;
  if (tile >= nTiles) return;          // wave-uniform
  int nodeBase = tile * 16;

  int m = lane & 15;
  int khalf = lane >> 4;
  int n = m;
  int node = nodeBase + m;

  // 8 A fragments (K=256 in 8 steps of 32): fused bias + relu + f16 cvt,
  // each half-row fetched as two b128 loads.
  const float4* ar = (const float4*)(outac + (size_t)node * GDIM);
  const float4* br = (const float4*)gat_bias;
  v16h A[8];
#pragma unroll
  for (int ks = 0; ks < 8; ++ks) {
    int b0 = ks * 8 + khalf * 2;        // k = ks*32 + khalf*8
    int b1 = ks * 8 + 4 + khalf * 2;    // k = ks*32 + 16 + khalf*8
    pack4_relu<0>(A[ks], ar[b0], br[b0]);
    pack4_relu<4>(A[ks], ar[b0 + 1], br[b0 + 1]);
    pack4_relu<8>(A[ks], ar[b1], br[b1]);
    pack4_relu<12>(A[ks], ar[b1 + 1], br[b1 + 1]);
  }

  for (int ct = 0; ct < FDIM / 16; ++ct) {
    int col0 = ct * 16;
    v8f c;
#pragma unroll
    for (int r = 0; r < 8; ++r) c[r] = 0.f;
#pragma unroll
    for (int ks = 0; ks < 8; ++ks) {
      v16h b = *(const v16h*)&ldsb[(((ct * 8) + ks) * 32 + lane) * 8];
      c = __builtin_amdgcn_wmma_f32_16x16x32_f16(false, A[ks], false, b,
                                                 (short)0, c, false, false);
    }
    float fcb = fc_b[col0 + n];
#pragma unroll
    for (int r = 0; r < 8; ++r)
      y[(size_t)(nodeBase + r + khalf * 8) * FDIM + col0 + n] = c[r] + fcb;
  }
}

// scalar tail for the last N%16 nodes (not launched when N%16==0)
__global__ void final_tail_kernel(const float* __restrict__ outac,
                                  const float* __restrict__ gat_bias,
                                  const float* __restrict__ fc_w,
                                  const float* __restrict__ fc_b,
                                  float* __restrict__ y, int startNode, int N) {
  int idx = blockIdx.x * blockDim.x + threadIdx.x;
  int total = (N - startNode) * FDIM;
  if (idx >= total) return;
  int node = startNode + idx / FDIM;
  int col = idx % FDIM;
  float acc = fc_b[col];
  for (int k = 0; k < GDIM; ++k) {
    float v = outac[(size_t)node * GDIM + k] + gat_bias[k];
    v = v > 0.f ? v : 0.f;
    acc += v * fc_w[k * FDIM + col];
  }
  y[(size_t)node * FDIM + col] = acc;
}

// ---------------------------------------------------------------- launcher
extern "C" void kernel_launch(void* const* d_in, const int* in_sizes, int n_in,
                              void* d_out, int out_size, void* d_ws, size_t ws_size,
                              hipStream_t stream) {
  const float* x        = (const float*)d_in[0];
  const int*   ei       = (const int*)d_in[1];
  const float* ea       = (const float*)d_in[2];
  const float* ecc_root = (const float*)d_in[3];
  const float* ecc_bias = (const float*)d_in[4];
  const float* mlp_w    = (const float*)d_in[5];
  const float* mlp_b    = (const float*)d_in[6];
  const float* gat_lin  = (const float*)d_in[7];
  const float* att_src  = (const float*)d_in[8];
  const float* att_dst  = (const float*)d_in[9];
  const float* gat_bias = (const float*)d_in[10];
  const float* fc_w     = (const float*)d_in[11];
  const float* fc_b     = (const float*)d_in[12];
  float* y = (float*)d_out;

  int N = in_sizes[0] / IN_C;   // 50000
  int E = in_sizes[1] / 2;      // 800000

  // workspace layout (floats): [agg | out_acc | denom | h | g | asrc | adst | emax]
  float* ws    = (float*)d_ws;
  float* agg   = ws;                               // N*16  (zeroed)
  float* outac = agg   + (size_t)N * OUT_C;        // N*256 (zeroed)
  float* denom = outac + (size_t)N * GDIM;         // N*4   (zeroed)
  float* hbuf  = denom + (size_t)N * HEADS;        // N*16
  float* g     = hbuf  + (size_t)N * OUT_C;        // N*256
  float* asrc  = g     + (size_t)N * GDIM;         // N*4
  float* adst  = asrc  + (size_t)N * HEADS;        // N*4
  float* emax  = adst  + (size_t)N * HEADS;        // N*4   (init -1e30)

  size_t zbytes = (size_t)N * (OUT_C + GDIM + HEADS) * sizeof(float);
  (void)hipMemsetAsync(agg, 0, zbytes, stream);
  init_emax_kernel<<<(N * HEADS + 255) / 256, 256, 0, stream>>>(emax, N * HEADS);

  ecc_edge_kernel<<<(E + 255) / 256, 256, 0, stream>>>(x, ei, ea, mlp_w, mlp_b, agg, E);
  ecc_node_kernel<<<(N + 255) / 256, 256, 0, stream>>>(x, ecc_root, ecc_bias, agg, hbuf, N);

  int nTiles = N / 16;              // full 16-node tiles for the WMMA kernels
  int startTail = nTiles * 16;
  int tail = N - startTail;
  int wblocks = (nTiles + 7) / 8;   // 8 waves per 256-thread block

  if (nTiles > 0)
    gat_gemm_kernel<<<wblocks, 256, 0, stream>>>(hbuf, gat_lin, g, nTiles);
  if (tail > 0)
    gat_tail_kernel<<<(tail * GDIM + 255) / 256, 256, 0, stream>>>(
        hbuf, gat_lin, g, startTail, N);

  att_score_kernel<<<(N * HEADS + 255) / 256, 256, 0, stream>>>(g, att_src, att_dst, asrc, adst, N);
  edge_max_kernel<<<(E + 255) / 256, 256, 0, stream>>>(ei, asrc, adst, emax, E);
  edge_denom_kernel<<<(E + 255) / 256, 256, 0, stream>>>(ei, asrc, adst, emax, denom, E);

  long long accT = (long long)E * 64;
  edge_accum_kernel<<<(int)((accT + 255) / 256), 256, 0, stream>>>(
      ei, asrc, adst, emax, denom, g, outac, E);

  if (nTiles > 0)
    final_gemm_kernel<<<wblocks, 256, 0, stream>>>(outac, gat_bias, fc_w, fc_b, y, nTiles);
  if (tail > 0)
    final_tail_kernel<<<(tail * FDIM + 255) / 256, 256, 0, stream>>>(
        outac, gat_bias, fc_w, fc_b, y, startTail, N);
}
